// OA_3_33097017983620
// MI455X (gfx1250) — matrix operand order repaired and checked by reading, not compiled, not run on hardware
//
#include <hip/hip_runtime.h>
#include <hip/hip_bf16.h>

typedef __attribute__((ext_vector_type(16))) _Float16 v16h;
typedef __attribute__((ext_vector_type(8)))  _Float16 v8h;
typedef __attribute__((ext_vector_type(8)))  float    v8f;

static constexpr int BB  = 8;
static constexpr int CC  = 256;
static constexpr int CKK = 64;
static constexpr int NN  = 2048;

// ---------------- WMMA fragment helpers (CDNA5 16x16x32 f16, wave32) -------
// A fragment: rows of a row-major [16][K] tile, k-contiguous.
//   lane L: m = L%16; k = 16*(L>=16) .. +15 contiguous (matches B fragment)
__device__ __forceinline__ v16h load_frag_a(const _Float16* tile, int ld) {
  const int lane = threadIdx.x & 31;
  return *(const v16h*)(tile + (size_t)(lane & 15) * ld + ((lane >> 4) & 1) * 16);
}
// B fragment: rows of a row-major [16][K] array whose row index is the WMMA
// column n'.  lane L: n' = L%16, k = 16*(L>=16) .. +15 contiguous.
__device__ __forceinline__ v16h load_frag_b(const _Float16* tile, int ld) {
  const int lane = threadIdx.x & 31;
  return *(const v16h*)(tile + (size_t)(lane & 15) * ld + ((lane >> 4) & 1) * 16);
}
__device__ __forceinline__ v8f wmma_f16(v16h a, v16h b, v8f c) {
  return __builtin_amdgcn_wmma_f32_16x16x32_f16(false, a, false, b,
                                                (short)0, c, false, false);
}

// ---------------- prep kernels ---------------------------------------------
__global__ void k_zero(float* colsum, float* bnsum, float* bnsq) {
  int i = blockIdx.x * blockDim.x + threadIdx.x;
  if (i < BB * NN) colsum[i] = 0.f;
  if (i < CC) { bnsum[i] = 0.f; bnsq[i] = 0.f; }
}

__global__ void k_cvt(const float* __restrict__ s, _Float16* __restrict__ d, int n) {
  int i = blockIdx.x * blockDim.x + threadIdx.x;
  if (i < n) d[i] = (_Float16)s[i];
}

// src [B][Cd][N] f32  ->  dst [B][N][Cd] f16
__global__ void k_transpose_cvt(const float* __restrict__ src,
                                _Float16* __restrict__ dst, int Cd) {
  size_t i = blockIdx.x * (size_t)blockDim.x + threadIdx.x;
  size_t total = (size_t)BB * Cd * NN;
  if (i >= total) return;
  int n = (int)(i % NN);
  size_t t = i / NN;
  int c = (int)(t % Cd);
  int b = (int)(t / Cd);
  dst[((size_t)b * NN + n) * Cd + c] = (_Float16)src[i];
}

// ---------------- projection GEMMs -----------------------------------------
// OutT[b][n][o] = sum_k X[b][n][k] * W[o][k]    (A = X rows, B = W rows)
__global__ void __launch_bounds__(256)
k_proj_xw(const _Float16* __restrict__ X, const _Float16* __restrict__ W,
          _Float16* __restrict__ OutT, int Kd, int Od) {
  const int b = blockIdx.y, n0 = blockIdx.x * 16;
  const int wave = threadIdx.x >> 5, lane = threadIdx.x & 31;
  const int col = lane & 15, hi = (lane >> 4) & 1;
  const _Float16* Xb = X + ((size_t)b * NN + n0) * Kd;
  const int numOT = Od >> 4;
  for (int ot = wave; ot < numOT; ot += 8) {
    v8f acc = {};
    for (int k0 = 0; k0 < Kd; k0 += 32)
      acc = wmma_f16(load_frag_a(Xb + k0, Kd),
                     load_frag_b(W + (size_t)ot * 16 * Kd + k0, Kd), acc);
    _Float16* outp = OutT + ((size_t)b * NN + n0 + hi * 8) * Od + ot * 16 + col;
#pragma unroll
    for (int r = 0; r < 8; ++r) outp[(size_t)r * Od] = (_Float16)acc[r];
  }
}

// V[b][o][n] = sum_c Wv[o][c] * qT[b][n][c] + bv[o]   (A = Wv rows, B = qT rows)
__global__ void __launch_bounds__(256)
k_proj_v(const _Float16* __restrict__ Wv, const _Float16* __restrict__ qT,
         const float* __restrict__ bv, _Float16* __restrict__ V) {
  const int b = blockIdx.y, n0 = blockIdx.x * 16;
  const int wave = threadIdx.x >> 5, lane = threadIdx.x & 31;
  const int col = lane & 15, hi = (lane >> 4) & 1;
  const _Float16* Qb = qT + ((size_t)b * NN + n0) * CC;
  for (int ot = wave; ot < 16; ot += 8) {
    v8f acc = {};
#pragma unroll
    for (int kk = 0; kk < 8; ++kk)
      acc = wmma_f16(load_frag_a(Wv + (size_t)ot * 16 * CC + kk * 32, CC),
                     load_frag_b(Qb + kk * 32, CC), acc);
#pragma unroll
    for (int r = 0; r < 8; ++r) {
      int o = ot * 16 + hi * 8 + r;
      V[((size_t)b * CC + o) * NN + n0 + col] = (_Float16)(acc[r] + bv[o]);
    }
  }
}

// ---------------- energy + softmax + transpose + colsum --------------------
// One workgroup per (b, 32-row nq block).  energy block [32][2048] f32 in LDS
// (256 KB of the 320 KB WGP LDS).  Each wave caches 2x8 A fragments so every
// B fragment feeds two WMMAs (A traffic amortized, ~32 flops/byte from L2).
__global__ void __launch_bounds__(256)
k_energy_softmax(const _Float16* __restrict__ AqT, const _Float16* __restrict__ KkT,
                 _Float16* __restrict__ attnT, float* __restrict__ colsum) {
  extern __shared__ float lds[];
  float* eblk = lds;                 // [32][NN]
  float* red = lds + 32 * NN;        // [256]
  float* rowstat = red + 256;        // [32]
  const int b = blockIdx.y, nq0 = blockIdx.x * 32;
  const int tid = threadIdx.x, wave = tid >> 5, lane = tid & 31;
  const int col = lane & 15, hi = (lane >> 4) & 1;

  // ---- energy: each wave owns 256 nk columns (16 tiles), K = C = 256 ----
  const _Float16* Ab = AqT + ((size_t)b * NN + nq0) * CC;
  v16h afrag[2][8];
#pragma unroll
  for (int qq = 0; qq < 2; ++qq)
#pragma unroll
    for (int kk = 0; kk < 8; ++kk)
      afrag[qq][kk] = load_frag_a(Ab + (size_t)qq * 16 * CC + kk * 32, CC);

  for (int t = 0; t < 16; ++t) {
    int nk0 = wave * 256 + t * 16;
    const _Float16* Kb = KkT + ((size_t)b * NN + nk0) * CC;
    v8f acc0 = {}, acc1 = {};
#pragma unroll
    for (int kk = 0; kk < 8; ++kk) {
      v16h bf = load_frag_b(Kb + kk * 32, CC);
      acc0 = wmma_f16(afrag[0][kk], bf, acc0);
      acc1 = wmma_f16(afrag[1][kk], bf, acc1);
    }
#pragma unroll
    for (int r = 0; r < 8; ++r) {
      eblk[(size_t)(hi * 8 + r) * NN + nk0 + col] = acc0[r];
      eblk[(size_t)(16 + hi * 8 + r) * NN + nk0 + col] = acc1[r];
    }
  }
  __syncthreads();

  // ---- row softmax: 8 threads per row (32 rows) ----
  const int row = tid >> 3, j = tid & 7;
  float* erow = eblk + (size_t)row * NN;
  float lmax = -3.0e38f;
  for (int nk = j; nk < NN; nk += 8) lmax = fmaxf(lmax, erow[nk]);
  red[tid] = lmax;
  __syncthreads();
  if (j == 0) {
    float m = red[row * 8];
    for (int t = 1; t < 8; ++t) m = fmaxf(m, red[row * 8 + t]);
    rowstat[row] = m;
  }
  __syncthreads();
  float rmax = rowstat[row], lsum = 0.f;
  for (int nk = j; nk < NN; nk += 8) {
    float e = __expf(erow[nk] - rmax);
    erow[nk] = e;
    lsum += e;
  }
  red[tid] = lsum;
  __syncthreads();
  if (j == 0) {
    float s = 0.f;
    for (int t = 0; t < 8; ++t) s += red[row * 8 + t];
    rowstat[row] = 1.f / s;
  }
  __syncthreads();

  // ---- transpose out of LDS (contiguous 64B stores) + colsum ----
  for (int nk = tid; nk < NN; nk += 256) {
    float cs = 0.f;
    v16h p0, p1;
#pragma unroll
    for (int m = 0; m < 16; ++m) {
      float pv = eblk[(size_t)m * NN + nk] * rowstat[m];
      cs += pv;
      p0[m] = (_Float16)pv;
    }
#pragma unroll
    for (int m = 0; m < 16; ++m) {
      float pv = eblk[(size_t)(16 + m) * NN + nk] * rowstat[16 + m];
      cs += pv;
      p1[m] = (_Float16)pv;
    }
    atomicAdd(&colsum[(size_t)b * NN + nk], cs);
    _Float16* dst = attnT + ((size_t)b * NN + nk) * NN + nq0;
    *(v16h*)dst = p0;
    *(v16h*)(dst + 16) = p1;
  }
}

// ---------------- x_r = V * attn, with L1 column renorm --------------------
// 4x4 register-blocked: each wave owns a 64(nk) x 64(c) tile; per K-step it
// loads 4 A + 4 B fragments and issues 16 WMMAs (32 flops/byte from L2).
// xrT[b][nk][c] = (sum_nq attnT[nk][nq] * V[c][nq]) / (1e-9 + colsum[nk])
__global__ void __launch_bounds__(256)
k_xr(const _Float16* __restrict__ attnT, const _Float16* __restrict__ V,
     const float* __restrict__ colsum, _Float16* __restrict__ xrT) {
  const int b = blockIdx.y;
  const int wave = threadIdx.x >> 5, lane = threadIdx.x & 31;
  const int col = lane & 15, hi = (lane >> 4) & 1;
  const int nk0 = blockIdx.x * 128 + (wave >> 2) * 64;  // 2 waves along nk
  const int c0 = (wave & 3) * 64;                       // 4 waves cover all C
  const _Float16* Ab = attnT + ((size_t)b * NN + nk0) * NN;
  const _Float16* Vb = V + ((size_t)b * CC + c0) * NN;
  const v8f vzero = {};
  v8f acc[4][4];
#pragma unroll
  for (int i = 0; i < 4; ++i)
#pragma unroll
    for (int jj = 0; jj < 4; ++jj) acc[i][jj] = vzero;

  for (int k0 = 0; k0 < NN; k0 += 32) {
    if (k0 + 256 < NN) __builtin_prefetch(Ab + k0 + 256, 0, 1);
    v16h a[4], bf[4];
#pragma unroll
    for (int i = 0; i < 4; ++i) a[i] = load_frag_a(Ab + (size_t)i * 16 * NN + k0, NN);
#pragma unroll
    for (int jj = 0; jj < 4; ++jj) bf[jj] = load_frag_b(Vb + (size_t)jj * 16 * NN + k0, NN);
#pragma unroll
    for (int i = 0; i < 4; ++i)
#pragma unroll
      for (int jj = 0; jj < 4; ++jj) acc[i][jj] = wmma_f16(a[i], bf[jj], acc[i][jj]);
  }

#pragma unroll
  for (int i = 0; i < 4; ++i) {
#pragma unroll
    for (int r = 0; r < 8; ++r) {
      int nk = nk0 + i * 16 + hi * 8 + r;
      float rcp = 1.f / (1e-9f + colsum[(size_t)b * NN + nk]);
      _Float16* dst = xrT + ((size_t)b * NN + nk) * CC + c0 + col;
#pragma unroll
      for (int jj = 0; jj < 4; ++jj) dst[jj * 16] = (_Float16)(acc[i][jj][r] * rcp);
    }
  }
}

// ---------------- y = Wt * x_r + bt, with BN stat accumulation -------------
// Same 4x4 register blocking, K = 256.
__global__ void __launch_bounds__(256)
k_final(const _Float16* __restrict__ xrT, const _Float16* __restrict__ Wt,
        const float* __restrict__ bt, float* __restrict__ y,
        float* __restrict__ bnsum, float* __restrict__ bnsq) {
  const int b = blockIdx.y;
  const int wave = threadIdx.x >> 5, lane = threadIdx.x & 31;
  const int col = lane & 15, hi = (lane >> 4) & 1;
  const int n0 = blockIdx.x * 128 + (wave >> 2) * 64;
  const int o0 = (wave & 3) * 64;
  const _Float16* Xb = xrT + ((size_t)b * NN + n0) * CC;
  const _Float16* Wb = Wt + (size_t)o0 * CC;
  const v8f vzero = {};
  v8f acc[4][4];
#pragma unroll
  for (int i = 0; i < 4; ++i)
#pragma unroll
    for (int jj = 0; jj < 4; ++jj) acc[i][jj] = vzero;

#pragma unroll
  for (int kk = 0; kk < 8; ++kk) {
    v16h a[4], bf[4];
#pragma unroll
    for (int i = 0; i < 4; ++i) a[i] = load_frag_a(Xb + (size_t)i * 16 * CC + kk * 32, CC);
#pragma unroll
    for (int jj = 0; jj < 4; ++jj) bf[jj] = load_frag_b(Wb + (size_t)jj * 16 * CC + kk * 32, CC);
#pragma unroll
    for (int i = 0; i < 4; ++i)
#pragma unroll
      for (int jj = 0; jj < 4; ++jj) acc[i][jj] = wmma_f16(a[i], bf[jj], acc[i][jj]);
  }

#pragma unroll
  for (int jj = 0; jj < 4; ++jj) {
    int o = o0 + jj * 16 + col;
    float bias = bt[o];
    float s = 0.f, s2 = 0.f;
#pragma unroll
    for (int i = 0; i < 4; ++i) {
      float* yp = y + ((size_t)b * CC + o) * NN + n0 + i * 16 + hi * 8;
#pragma unroll
      for (int r = 0; r < 8; ++r) {
        float v = acc[i][jj][r] + bias;
        yp[r] = v;
        s += v;
        s2 += v * v;
      }
    }
    atomicAdd(&bnsum[o], s);
    atomicAdd(&bnsq[o], s2);
  }
}

// ---------------- batch-norm finalize + apply ------------------------------
__global__ void k_bn_final(const float* __restrict__ bnsum, const float* __restrict__ bnsq,
                           const float* __restrict__ gamma, const float* __restrict__ beta,
                           float* __restrict__ scale, float* __restrict__ shift) {
  int c = threadIdx.x;
  const float inv = 1.f / (float)(BB * NN);
  float mean = bnsum[c] * inv;
  float var = bnsq[c] * inv - mean * mean;
  float sc = gamma[c] * rsqrtf(var + 1e-5f);
  scale[c] = sc;
  shift[c] = beta[c] - mean * sc;
}

__global__ void k_bn_apply(const float* __restrict__ y, const float* __restrict__ scale,
                           const float* __restrict__ shift, float* __restrict__ out) {
  size_t i = blockIdx.x * (size_t)blockDim.x + threadIdx.x;
  if (i >= (size_t)BB * CC * NN) return;
  int c = (int)((i / NN) % CC);
  float v = y[i] * scale[c] + shift[c];
  out[i] = v > 0.f ? v : 0.f;
}

// ---------------- host launcher --------------------------------------------
extern "C" void kernel_launch(void* const* d_in, const int* in_sizes, int n_in,
                              void* d_out, int out_size, void* d_ws, size_t ws_size,
                              hipStream_t stream) {
  (void)in_sizes; (void)n_in; (void)out_size; (void)ws_size;
  const float* q    = (const float*)d_in[0];
  const float* x    = (const float*)d_in[1];
  const float* Wq   = (const float*)d_in[2];
  const float* Wk   = (const float*)d_in[3];
  const float* Wv   = (const float*)d_in[4];
  const float* bv   = (const float*)d_in[5];
  const float* Wt   = (const float*)d_in[6];
  const float* bt   = (const float*)d_in[7];
  const float* gmm  = (const float*)d_in[8];
  const float* beta = (const float*)d_in[9];
  float* out = (float*)d_out;

  char* p = (char*)d_ws;
  auto take = [&](size_t bytes) -> char* {
    char* r = p;
    p += (bytes + 255) & ~(size_t)255;
    return r;
  };
  _Float16* qT    = (_Float16*)take((size_t)BB * NN * CC * 2);
  _Float16* xT    = (_Float16*)take((size_t)BB * NN * CKK * 2);
  _Float16* Wq16  = (_Float16*)take((size_t)CC * CC * 2);
  _Float16* Wk16  = (_Float16*)take((size_t)CC * CKK * 2);
  _Float16* Wv16  = (_Float16*)take((size_t)CC * CC * 2);
  _Float16* Wt16  = (_Float16*)take((size_t)CC * CC * 2);
  _Float16* AqT   = (_Float16*)take((size_t)BB * NN * CC * 2);
  _Float16* KkT   = (_Float16*)take((size_t)BB * NN * CC * 2);
  _Float16* V16   = (_Float16*)take((size_t)BB * CC * NN * 2);
  _Float16* attnT = (_Float16*)take((size_t)BB * NN * NN * 2);
  _Float16* xrT   = (_Float16*)take((size_t)BB * NN * CC * 2);
  float* colsum   = (float*)take((size_t)BB * NN * 4);
  float* y32      = (float*)take((size_t)BB * CC * NN * 4);
  float* bnsum    = (float*)take(CC * 4);
  float* bnsq     = (float*)take(CC * 4);
  float* scale    = (float*)take(CC * 4);
  float* shift    = (float*)take(CC * 4);

  const dim3 blk(256);
  k_zero<<<dim3((BB * NN + 255) / 256), blk, 0, stream>>>(colsum, bnsum, bnsq);

  k_cvt<<<dim3((CC * CC + 255) / 256), blk, 0, stream>>>(Wq, Wq16, CC * CC);
  k_cvt<<<dim3((CC * CKK + 255) / 256), blk, 0, stream>>>(Wk, Wk16, CC * CKK);
  k_cvt<<<dim3((CC * CC + 255) / 256), blk, 0, stream>>>(Wv, Wv16, CC * CC);
  k_cvt<<<dim3((CC * CC + 255) / 256), blk, 0, stream>>>(Wt, Wt16, CC * CC);
  k_transpose_cvt<<<dim3((unsigned)(((size_t)BB * CC * NN + 255) / 256)), blk, 0, stream>>>(q, qT, CC);
  k_transpose_cvt<<<dim3((unsigned)(((size_t)BB * CKK * NN + 255) / 256)), blk, 0, stream>>>(x, xT, CKK);

  const dim3 grd16(NN / 16, BB);
  k_proj_xw<<<grd16, blk, 0, stream>>>(qT, Wq16, AqT, CC, CC);
  k_proj_xw<<<grd16, blk, 0, stream>>>(xT, Wk16, KkT, CKK, CC);
  k_proj_v<<<grd16, blk, 0, stream>>>(Wv16, qT, bv, V16);

  size_t smem = (size_t)(32 * NN + 256 + 32) * sizeof(float);  // ~257 KB
  k_energy_softmax<<<dim3(NN / 32, BB), blk, smem, stream>>>(AqT, KkT, attnT, colsum);
  k_xr<<<dim3(NN / 128, BB), blk, 0, stream>>>(attnT, V16, colsum, xrT);
  k_final<<<dim3(NN / 128, BB), blk, 0, stream>>>(xrT, Wt16, bt, y32, bnsum, bnsq);
  k_bn_final<<<dim3(1), blk, 0, stream>>>(bnsum, bnsq, gmm, beta, scale, shift);
  k_bn_apply<<<dim3((unsigned)(((size_t)BB * CC * NN + 255) / 256)), blk, 0, stream>>>(y32, scale, shift, out);
}
